// MoEFFN_82257213653365
// MI455X (gfx1250) — matrix operand order repaired and checked by reading, not compiled, not run on hardware
//
#include <hip/hip_runtime.h>

// ---------------- problem dims ----------------
#define H_DIM 1024
#define E_NUM 8
#define F_DIM 4096
#define N_TOK 8192

// ---------------- vector types ----------------
typedef __attribute__((ext_vector_type(16))) __bf16       v16bf;
typedef __attribute__((ext_vector_type(8)))  float        v8f;
typedef __attribute__((ext_vector_type(4)))  float        v4f;
typedef __attribute__((ext_vector_type(4)))  unsigned int v4u;

union FragBF { v16bf bf; v4u u[2]; };

static __device__ __forceinline__ v8f wmma_bf16(v16bf a, v16bf b, v8f c) {
  return __builtin_amdgcn_wmma_f32_16x16x32_bf16(
      false, a, false, b, (short)0, c, false, false);
}

static __device__ __forceinline__ float gelu_exact(float v) {
  return 0.5f * v * (1.0f + erff(v * 0.70710678118654752f));
}

// LDS byte offset of a shared-memory pointer: flat-LDS aperture keeps the
// LDS address in bits [31:0] (CDNA5 ISA 10.2), so truncation is exact.
static __device__ __forceinline__ unsigned lds_off(const void* p) {
  return (unsigned)(size_t)p;
}

// async global -> LDS, 16B/lane. INST_OFFSET is applied to BOTH the LDS
// destination and the global source (CDNA5 ISA 08 §4.4), and the intra-stage
// segment offset is identical on both sides, so it rides in the immediate.
static __device__ __forceinline__ void async_b128(unsigned ldsoff, const __bf16* g, int imm) {
  asm volatile("global_load_async_to_lds_b128 %0, %1, off offset:%2"
               :: "v"(ldsoff), "v"(g), "i"(imm) : "memory");
}
#define WAIT_ASYNC_PREV() asm volatile("s_wait_asynccnt 0x4" ::: "memory")
#define WAIT_ASYNC_ALL()  asm volatile("s_wait_asynccnt 0x0" ::: "memory")

// ---------------- x fp32 -> bf16 ----------------
__global__ __launch_bounds__(256) void moe_cvt_x(const float* __restrict__ x,
                                                 __bf16* __restrict__ xb) {
  size_t i = ((size_t)blockIdx.x * 256 + threadIdx.x) * 8;
  v4f a = *(const v4f*)(x + i);
  v4f b = *(const v4f*)(x + i + 4);
  union { __bf16 h[8]; v4u u; } o;
#pragma unroll
  for (int j = 0; j < 4; ++j) { o.h[j] = (__bf16)a[j]; o.h[4 + j] = (__bf16)b[j]; }
  *(v4u*)(xb + i) = o.u;
}

// ---------------- weight convert + transpose: W[K][N] f32 -> WT[N][K] bf16 ----
__global__ __launch_bounds__(256) void moe_transpose_cvt(const float* __restrict__ W,
                                                         __bf16* __restrict__ WT,
                                                         int K, int N) {
  const int e = blockIdx.z;
  W  += (size_t)e * K * N;
  WT += (size_t)e * K * N;
  const int n0 = blockIdx.x * 64;
  const int k0 = blockIdx.y * 64;
  __shared__ __bf16 t[64 * 80];

  const int tid = threadIdx.x;
  {
    const int r    = tid >> 2;          // k within tile
    const int cseg = (tid & 3) * 16;    // n within tile
    const float* src = W + (size_t)(k0 + r) * N + n0 + cseg;
    union { __bf16 h[16]; v4u u[2]; } o;
#pragma unroll
    for (int q = 0; q < 4; ++q) {
      v4f v = *(const v4f*)(src + q * 4);
#pragma unroll
      for (int j = 0; j < 4; ++j) o.h[q * 4 + j] = (__bf16)v[j];
    }
    *(v4u*)(t + r * 80 + cseg)     = o.u[0];
    *(v4u*)(t + r * 80 + cseg + 8) = o.u[1];
  }
  __syncthreads();
  {
    const int n    = tid >> 2;          // n within tile
    const int kseg = (tid & 3) * 16;    // k within tile
    union { __bf16 h[16]; v4u u[2]; } o;
#pragma unroll
    for (int i = 0; i < 16; ++i) o.h[i] = t[(kseg + i) * 80 + n];
    __bf16* dst = WT + (size_t)(n0 + n) * K + k0 + kseg;
    *(v4u*)dst       = o.u[0];
    *(v4u*)(dst + 8) = o.u[1];
  }
}

// ---------------- gating: top-2 + softmax + routing ----------------
__global__ __launch_bounds__(256) void moe_gate(const float* __restrict__ x,
                                                const float* __restrict__ gw,
                                                int* __restrict__ cnt,
                                                int* __restrict__ tok,
                                                float* __restrict__ wts) {
  const int lane = threadIdx.x & 31;
  const int wave = threadIdx.x >> 5;
  const int token = blockIdx.x * 8 + wave;

  float acc[E_NUM];
#pragma unroll
  for (int e = 0; e < E_NUM; ++e) acc[e] = 0.0f;

  const float* xr = x + (size_t)token * H_DIM;
#pragma unroll 4
  for (int i = 0; i < H_DIM / 32; ++i) {
    const int h = i * 32 + lane;
    const float xv = xr[h];
    const float* g = gw + (size_t)h * E_NUM;
#pragma unroll
    for (int e = 0; e < E_NUM; ++e) acc[e] += xv * g[e];
  }
#pragma unroll
  for (int off = 16; off >= 1; off >>= 1) {
#pragma unroll
    for (int e = 0; e < E_NUM; ++e) acc[e] += __shfl_xor(acc[e], off, 32);
  }

  if (lane == 0) {
    float b0 = -3.4e38f, b1 = -3.4e38f;
    int i0 = 0, i1 = 0;
#pragma unroll
    for (int e = 0; e < E_NUM; ++e) {
      float l = acc[e];
      if (l > b0) { b1 = b0; i1 = i0; b0 = l; i0 = e; }
      else if (l > b1) { b1 = l; i1 = e; }
    }
    float e1 = expf(b1 - b0);
    float inv = 1.0f / (1.0f + e1);
    int p0 = atomicAdd(&cnt[i0], 1);
    tok[i0 * N_TOK + p0] = token; wts[i0 * N_TOK + p0] = inv;
    int p1 = atomicAdd(&cnt[i1], 1);
    tok[i1 * N_TOK + p1] = token; wts[i1 * N_TOK + p1] = e1 * inv;
  }
}

// ---------------- prefix sum over 8 expert counts ----------------
__global__ void moe_scan(const int* __restrict__ cnt, int* __restrict__ basep) {
  if (threadIdx.x == 0) {
    int b = 0;
#pragma unroll
    for (int e = 0; e < E_NUM; ++e) { basep[e] = b; b += cnt[e]; }
    basep[E_NUM] = b;
  }
}

// ======== shared GEMM core: async double-buffered staging + WMMA ========
// LDS: As/Bs = 2 x [128 rows x 40 bf16] (row stride 80B). Per stage each
// thread issues 4 async b128 ops (A:2, B:2); imm {0,16} selects the segment.
template <int KDIM>
static __device__ __forceinline__ void gemm_core(const __bf16* ag,
                                                 const __bf16* bg,
                                                 __bf16* As, __bf16* Bs,
                                                 unsigned a_off, unsigned b_off,
                                                 int lane, int wm, int wn,
                                                 v8f acc[4][2]) {
  constexpr int NIT2 = KDIM / 64;          // pairs of 32-wide k-stages
  const int rsel   = lane & 15;
  const int khalfA = (lane >> 4) * 8;
  const int khalfB = (lane >> 4) * 8;
  const unsigned a0 = a_off, a1 = a_off + 10240u;
  const unsigned b0 = b_off, b1 = b_off + 10240u;

  // prologue: stage 0 -> buffer 0
  async_b128(a0, ag, 0);  async_b128(a0, ag, 16);
  async_b128(b0, bg, 0);  async_b128(b0, bg, 16);
  const __bf16* agp = ag + 32;             // next stage base (64B per stage)
  const __bf16* bgp = bg + 32;

  auto compute = [&](const __bf16* Ab, const __bf16* Bb) {
    FragBF a[4], b[2];
#pragma unroll
    for (int i = 0; i < 4; ++i) {
      const int r = wm * 64 + i * 16 + rsel;
      a[i].u[0] = *(const v4u*)(Ab + r * 40 + khalfA);
      a[i].u[1] = *(const v4u*)(Ab + r * 40 + 16 + khalfA);
    }
#pragma unroll
    for (int j = 0; j < 2; ++j) {
      const int c = wn * 32 + j * 16 + rsel;
      b[j].u[0] = *(const v4u*)(Bb + c * 40 + khalfB);
      b[j].u[1] = *(const v4u*)(Bb + c * 40 + 16 + khalfB);
    }
#pragma unroll
    for (int i = 0; i < 4; ++i)
#pragma unroll
      for (int j = 0; j < 2; ++j)
        acc[i][j] = wmma_bf16(a[i].bf, b[j].bf, acc[i][j]);
  };

  for (int it2 = 0; it2 < NIT2; ++it2) {
    // issue stage 2*it2+1 -> buffer 1
    async_b128(a1, agp, 0);  async_b128(a1, agp, 16);
    async_b128(b1, bgp, 0);  async_b128(b1, bgp, 16);
    agp += 32;  bgp += 32;
    WAIT_ASYNC_PREV();                 // own stage 2*it2 complete (in-order)
    __syncthreads();                   // buffer 0 ready across all waves
    compute(As, Bs);
    __syncthreads();                   // all waves done reading buffer 0
    // issue stage 2*it2+2 -> buffer 0 (unless last)
    if (it2 + 1 < NIT2) {
      async_b128(a0, agp, 0);  async_b128(a0, agp, 16);
      async_b128(b0, bgp, 0);  async_b128(b0, bgp, 16);
      agp += 32;  bgp += 32;
      WAIT_ASYNC_PREV();               // own stage 2*it2+1 complete
    } else {
      WAIT_ASYNC_ALL();
    }
    __syncthreads();                   // buffer 1 ready across all waves
    compute(As + 5120, Bs + 5120);
    __syncthreads();                   // all waves done reading buffer 1
  }
}

// ---------------- GEMM1: h = gelu(x_g @ W1[e] + b1[e]) ----------------
__global__ __launch_bounds__(256) void moe_gemm1(const __bf16* __restrict__ xb,
                                                 const __bf16* __restrict__ W1T,
                                                 const float* __restrict__ b1,
                                                 const int* __restrict__ cnt,
                                                 const int* __restrict__ basep,
                                                 const int* __restrict__ tok,
                                                 __bf16* __restrict__ hout) {
  const int e  = blockIdx.z;
  const int n0 = blockIdx.x * 128;
  const int r0 = blockIdx.y * 128;
  const int cnte = cnt[e];
  if (r0 >= cnte) return;

  __shared__ __bf16 As[2 * 128 * 40];
  __shared__ __bf16 Bs[2 * 128 * 40];

  const int tid  = threadIdx.x;
  const int lane = tid & 31;
  const int wave = tid >> 5;
  const int wm = wave >> 2, wn = wave & 3;

  const int row = tid >> 1, seg = tid & 1;
  const int tIdx  = r0 + row;
  const int tokId = (tIdx < cnte) ? tok[e * N_TOK + tIdx] : 0;
  const __bf16* ag = xb + (size_t)tokId * H_DIM + seg * 16;
  const __bf16* bg = W1T + (size_t)e * F_DIM * H_DIM + (size_t)(n0 + row) * H_DIM + seg * 16;
  const unsigned a_off = lds_off(As) + row * 80u + seg * 32u;
  const unsigned b_off = lds_off(Bs) + row * 80u + seg * 32u;

  const v8f zero8 = {0.f, 0.f, 0.f, 0.f, 0.f, 0.f, 0.f, 0.f};
  v8f acc[4][2];
#pragma unroll
  for (int i = 0; i < 4; ++i)
#pragma unroll
    for (int j = 0; j < 2; ++j) acc[i][j] = zero8;

  gemm_core<H_DIM>(ag, bg, As, Bs, a_off, b_off, lane, wm, wn, acc);

  const int hb = basep[e];
  const int nlane = lane & 15;
  const int mhalf = (lane >> 4) * 8;
  float bias[2];
#pragma unroll
  for (int j = 0; j < 2; ++j)
    bias[j] = b1[e * F_DIM + n0 + wn * 32 + j * 16 + nlane];
#pragma unroll
  for (int i = 0; i < 4; ++i)
#pragma unroll
    for (int j = 0; j < 2; ++j) {
      const int col = n0 + wn * 32 + j * 16 + nlane;
#pragma unroll
      for (int r = 0; r < 8; ++r) {
        const int rr = r0 + wm * 64 + i * 16 + mhalf + r;
        if (rr < cnte) {
          float v = acc[i][j][r] + bias[j];
          hout[(size_t)(hb + rr) * F_DIM + col] = (__bf16)gelu_exact(v);
        }
      }
    }
}

// ---------------- GEMM2: out[token] += w * (h @ W2[e] + b2[e]) ----------------
__global__ __launch_bounds__(256) void moe_gemm2(const __bf16* __restrict__ hbuf,
                                                 const __bf16* __restrict__ W2T,
                                                 const float* __restrict__ b2,
                                                 const int* __restrict__ cnt,
                                                 const int* __restrict__ basep,
                                                 const int* __restrict__ tok,
                                                 const float* __restrict__ wts,
                                                 float* __restrict__ out) {
  const int e  = blockIdx.z;
  const int n0 = blockIdx.x * 128;
  const int r0 = blockIdx.y * 128;
  const int cnte = cnt[e];
  if (r0 >= cnte) return;
  const int hb = basep[e];

  __shared__ __bf16 As[2 * 128 * 40];
  __shared__ __bf16 Bs[2 * 128 * 40];

  const int tid  = threadIdx.x;
  const int lane = tid & 31;
  const int wave = tid >> 5;
  const int wm = wave >> 2, wn = wave & 3;

  const int row = tid >> 1, seg = tid & 1;
  int rIdx = r0 + row;
  if (rIdx >= cnte) rIdx = cnte - 1;   // clamp; padding rows masked in epilogue
  const __bf16* ag = hbuf + (size_t)(hb + rIdx) * F_DIM + seg * 16;
  const __bf16* bg = W2T + (size_t)e * F_DIM * H_DIM + (size_t)(n0 + row) * F_DIM + seg * 16;
  const unsigned a_off = lds_off(As) + row * 80u + seg * 32u;
  const unsigned b_off = lds_off(Bs) + row * 80u + seg * 32u;

  const v8f zero8 = {0.f, 0.f, 0.f, 0.f, 0.f, 0.f, 0.f, 0.f};
  v8f acc[4][2];
#pragma unroll
  for (int i = 0; i < 4; ++i)
#pragma unroll
    for (int j = 0; j < 2; ++j) acc[i][j] = zero8;

  gemm_core<F_DIM>(ag, bg, As, Bs, a_off, b_off, lane, wm, wn, acc);

  const int nlane = lane & 15;
  const int mhalf = (lane >> 4) * 8;
  float bias[2];
#pragma unroll
  for (int j = 0; j < 2; ++j)
    bias[j] = b2[e * H_DIM + n0 + wn * 32 + j * 16 + nlane];
#pragma unroll
  for (int i = 0; i < 4; ++i)
#pragma unroll
    for (int r = 0; r < 8; ++r) {
      const int rr = r0 + wm * 64 + i * 16 + mhalf + r;
      if (rr < cnte) {
        const int   t = tok[e * N_TOK + rr];
        const float w = wts[e * N_TOK + rr];
#pragma unroll
        for (int j = 0; j < 2; ++j) {
          const int col = n0 + wn * 32 + j * 16 + nlane;
          atomicAdd(&out[(size_t)t * H_DIM + col], (acc[i][j][r] + bias[j]) * w);
        }
      }
    }
}

// ---------------- launcher ----------------
extern "C" void kernel_launch(void* const* d_in, const int* in_sizes, int n_in,
                              void* d_out, int out_size, void* d_ws, size_t ws_size,
                              hipStream_t stream) {
  const float* x  = (const float*)d_in[0];
  const float* gw = (const float*)d_in[1];
  const float* W1 = (const float*)d_in[2];
  const float* b1 = (const float*)d_in[3];
  const float* W2 = (const float*)d_in[4];
  const float* b2 = (const float*)d_in[5];
  float* out = (float*)d_out;

  char* ws = (char*)d_ws;
  int*    cnt   = (int*)(ws + 0);
  int*    basep = (int*)(ws + 256);
  int*    tokp  = (int*)(ws + 512);
  float*  wtsp  = (float*)(ws + 512 + (size_t)E_NUM * N_TOK * 4);
  size_t  off   = 525312;
  __bf16* xb    = (__bf16*)(ws + off);  off += (size_t)N_TOK * H_DIM * 2;         // 16 MB
  __bf16* hbuf  = (__bf16*)(ws + off);  off += (size_t)2 * N_TOK * F_DIM * 2;     // 128 MB
  __bf16* W1T   = (__bf16*)(ws + off);  off += (size_t)E_NUM * H_DIM * F_DIM * 2; // 64 MB
  __bf16* W2T   = (__bf16*)(ws + off);                                            // 64 MB

  hipMemsetAsync(cnt, 0, E_NUM * sizeof(int), stream);
  hipMemsetAsync(out, 0, (size_t)out_size * sizeof(float), stream);

  moe_cvt_x<<<(N_TOK * H_DIM) / (256 * 8), 256, 0, stream>>>(x, xb);
  moe_gate <<<N_TOK / 8, 256, 0, stream>>>(x, gw, cnt, tokp, wtsp);
  moe_scan <<<1, 32, 0, stream>>>(cnt, basep);
  moe_transpose_cvt<<<dim3(F_DIM / 64, H_DIM / 64, E_NUM), 256, 0, stream>>>(W1, W1T, H_DIM, F_DIM);
  moe_transpose_cvt<<<dim3(H_DIM / 64, F_DIM / 64, E_NUM), 256, 0, stream>>>(W2, W2T, F_DIM, H_DIM);
  moe_gemm1<<<dim3(F_DIM / 128, N_TOK / 128, E_NUM), 256, 0, stream>>>(
      xb, W1T, b1, cnt, basep, tokp, hbuf);
  moe_gemm2<<<dim3(H_DIM / 128, N_TOK / 128, E_NUM), 256, 0, stream>>>(
      hbuf, W2T, b2, cnt, basep, tokp, wtsp, out);
}
